// Attention_16243566313816
// MI455X (gfx1250) — matrix operand order repaired and checked by reading, not compiled, not run on hardware
//
#include <hip/hip_runtime.h>

// ---------------- problem constants ----------------
#define DIMC     256
#define HEADS    8
#define HDIM     32
#define HH       48
#define WWW      48
#define NN       (HH * WWW)          // 2304
#define BATCH    2
#define QSCALE   0.17677669529663687f   // 32^-0.5

typedef _Float16 v16h __attribute__((ext_vector_type(16)));
typedef float    v8f  __attribute__((ext_vector_type(8)));

__device__ __forceinline__ v8f wmma16(v16h a, v16h b, v8f c) {
    // D = A(16x32 f16) * B(32x16 f16) + C(16x16 f32)
    return __builtin_amdgcn_wmma_f32_16x16x32_f16(
        /*neg_a=*/false, a, /*neg_b=*/false, b,
        /*c_mod=*/(short)0, c, /*reuse_a=*/false, /*reuse_b=*/false);
}

// =====================================================================
// Kernel 1: qkv 1x1 conv GEMM.  qkv[b,o,n] = sum_c W[o,c] * x[b,c,n]
// One wave computes one 16(o) x 16(n) f32 tile, K=256 in 8 WMMA steps.
// Writes:
//   qT  f16 [bh][n][d]   (q pre-scaled by 1/sqrt(d))  -> score B-matrix
//   kT  f16 [bh][n][d]                                -> score A-matrix
//   vR  f16 [bh][d][n]                                -> PV A-matrix
//   vF  f32 [b][c][n]                                 -> lepe input
// =====================================================================
__global__ void __launch_bounds__(128)
qkv_kernel(const float* __restrict__ x, const float* __restrict__ w,
           _Float16* __restrict__ qT, _Float16* __restrict__ kT,
           _Float16* __restrict__ vR, float* __restrict__ vF) {
    int tile  = blockIdx.x * 4 + (threadIdx.x >> 5);  // [0, 2*48*144)
    int ntile = tile % (NN / 16); tile /= (NN / 16);
    int otile = tile % 48;
    int b     = tile / 48;
    int lane  = threadIdx.x & 31;
    int colq  = lane & 15;
    bool lo   = lane < 16;

    const float* wrow = w + (size_t)(otile * 16 + colq) * DIMC;   // A row = lane%16
    const float* xb   = x + (size_t)b * DIMC * NN;
    int ncol = ntile * 16 + colq;

    v8f acc = {};
    for (int cb = 0; cb < DIMC; cb += 32) {
        int a0 = cb + (lo ? 0 : 16);
        int a1 = cb + (lo ? 8 : 24);
        v16h A;
#pragma unroll
        for (int j = 0; j < 8; ++j) {
            A[j]     = (_Float16)wrow[a0 + j];
            A[8 + j] = (_Float16)wrow[a1 + j];
        }
        int k0 = cb + (lo ? 0 : 16);
        v16h B;
#pragma unroll
        for (int j = 0; j < 16; ++j)
            B[j] = (_Float16)xb[(size_t)(k0 + j) * NN + ncol];
        acc = wmma16(A, B, acc);
    }

    int o0    = otile * 16;
    int hh    = o0 / 96;           // head
    int r0    = o0 % 96;           // position within [q|k|v] (3*32)
    int bh    = b * HEADS + hh;
    int mbase = lo ? 0 : 8;

    if (r0 < 32) {                 // Q (pre-scaled)
#pragma unroll
        for (int j = 0; j < 8; ++j) {
            int dd = r0 + mbase + j;
            qT[((size_t)bh * NN + ncol) * HDIM + dd] = (_Float16)(acc[j] * QSCALE);
        }
    } else if (r0 < 64) {          // K
#pragma unroll
        for (int j = 0; j < 8; ++j) {
            int dd = (r0 - 32) + mbase + j;
            kT[((size_t)bh * NN + ncol) * HDIM + dd] = (_Float16)acc[j];
        }
    } else {                       // V (f16 row-major + f32 spatial)
#pragma unroll
        for (int j = 0; j < 8; ++j) {
            int dd = (r0 - 64) + mbase + j;
            vR[((size_t)bh * HDIM + dd) * NN + ncol] = (_Float16)acc[j];
            vF[((size_t)b * DIMC + hh * HDIM + dd) * NN + ncol] = acc[j];
        }
    }
}

// =====================================================================
// Kernel 2: depthwise 3x3 conv (SAME, zero pad) + bias  ->  lepe
// =====================================================================
__global__ void __launch_bounds__(256)
lepe_kernel(const float* __restrict__ vF, const float* __restrict__ wl,
            const float* __restrict__ bl, float* __restrict__ lep) {
    int idx = blockIdx.x * blockDim.x + threadIdx.x;
    if (idx >= BATCH * DIMC * NN) return;
    int n = idx % NN;
    int c = (idx / NN) % DIMC;
    int y = n / WWW, xp = n % WWW;
    const float* vp = vF + (size_t)(idx - n);   // base of [b][c] plane
    const float* wp = wl + c * 9;
    float s = bl[c];
#pragma unroll
    for (int ky = -1; ky <= 1; ++ky) {
        int yy = y + ky;
        if (yy < 0 || yy >= HH) continue;
#pragma unroll
        for (int kx = -1; kx <= 1; ++kx) {
            int xx = xp + kx;
            if (xx < 0 || xx >= WWW) continue;
            s += vp[yy * WWW + xx] * wp[(ky + 1) * 3 + (kx + 1)];
        }
    }
    lep[idx] = s;
}

// =====================================================================
// Kernel 3: attention.  One wave owns one (b,h, 16-query tile).
// Single pass over keys in chunks of 32 with online (flash) softmax.
// S = K^T Q via 2 WMMAs, O += V P via 2 WMMAs per chunk.
// K/V per head = ~300 KB, reused by 144 waves -> lives in the 192MB L2.
// =====================================================================
__global__ void __launch_bounds__(128)
attn_kernel(const _Float16* __restrict__ qT, const _Float16* __restrict__ kT,
            const _Float16* __restrict__ vR, float* __restrict__ aO) {
    int wid   = blockIdx.x * 4 + (threadIdx.x >> 5);   // [0, 16*144)
    int qtile = wid % (NN / 16);
    int bh    = wid / (NN / 16);
    int lane  = threadIdx.x & 31;
    int colq  = lane & 15;
    bool lo   = lane < 16;
    int b     = bh / HEADS;
    int hh    = bh % HEADS;

    // Q as B-matrix: lane = column q, K(=d) = (lo?0:16)+c, contiguous in qT
    int qcol = qtile * 16 + colq;
    const _Float16* qp = qT + ((size_t)bh * NN + qcol) * HDIM + (lo ? 0 : 16);
    v16h Bq;
#pragma unroll
    for (int j = 0; j < 16; ++j) Bq[j] = qp[j];

    const _Float16* kbase = kT + (size_t)bh * NN * HDIM;
    const _Float16* vbase = vR + (size_t)bh * HDIM * NN;
    int arun0 = lo ? 0 : 16;
    int arun1 = lo ? 8 : 24;

    float m = -3.0e38f, lsum = 0.0f;
    v8f o0 = {}, o1 = {};

    for (int kb = 0; kb < NN; kb += 32) {
        // ---- A tiles of K^T: row = key index, K-dim = d (contiguous) ----
        const _Float16* k0 = kbase + (size_t)(kb + colq) * HDIM;
        const _Float16* k1 = kbase + (size_t)(kb + 16 + colq) * HDIM;
        v16h A0, A1;
#pragma unroll
        for (int j = 0; j < 8; ++j) {
            A0[j] = k0[arun0 + j]; A0[8 + j] = k0[arun1 + j];
            A1[j] = k1[arun0 + j]; A1[8 + j] = k1[arun1 + j];
        }
        v8f z = {};
        v8f S0 = wmma16(A0, Bq, z);
        v8f S1 = wmma16(A1, Bq, z);

        // prefetch next chunk of K (speculative; dropped if OOB)
        __builtin_prefetch((const void*)(k0 + 32 * HDIM), 0, 1);
        __builtin_prefetch((const void*)(k1 + 32 * HDIM), 0, 1);

        // ---- online softmax over the 32 keys of this chunk ----
        float tmax = -3.0e38f;
#pragma unroll
        for (int j = 0; j < 8; ++j) tmax = fmaxf(tmax, fmaxf(S0[j], S1[j]));
        tmax = fmaxf(tmax, __shfl_xor(tmax, 16, 32));   // lanes l,l+16 share a column
        float mnew = fmaxf(m, tmax);
        float sf   = __expf(m - mnew);
        m = mnew;
        lsum *= sf;
#pragma unroll
        for (int j = 0; j < 8; ++j) { o0[j] *= sf; o1[j] *= sf; }

        float p0[8], p1[8], psum = 0.0f;
#pragma unroll
        for (int j = 0; j < 8; ++j) {
            p0[j] = __expf(S0[j] - m);
            p1[j] = __expf(S1[j] - m);
            psum += p0[j] + p1[j];
        }
        psum += __shfl_xor(psum, 16, 32);
        lsum += psum;

        // ---- rebuild P (C/D layout) into f16 B-matrix layout ----
        // B comp c <-> key row (lo?0:16)+c; rows 8..15 live in the partner lane.
        v16h Bp;
#pragma unroll
        for (int j = 0; j < 8; ++j) {
            float x0 = __shfl_xor(p0[j], 16, 32);
            float x1 = __shfl_xor(p1[j], 16, 32);
            Bp[j]     = (_Float16)(lo ? p0[j] : x1);
            Bp[8 + j] = (_Float16)(lo ? x0 : p1[j]);
        }

        // ---- A tiles of V: row = d, K-dim = key chunk (contiguous) ----
        const _Float16* v0 = vbase + (size_t)colq * NN + kb;
        const _Float16* v1 = vbase + (size_t)(16 + colq) * NN + kb;
        v16h Av0, Av1;
#pragma unroll
        for (int j = 0; j < 8; ++j) {
            Av0[j] = v0[arun0 + j]; Av0[8 + j] = v0[arun1 + j];
            Av1[j] = v1[arun0 + j]; Av1[8 + j] = v1[arun1 + j];
        }
        __builtin_prefetch((const void*)(v0 + 32), 0, 1);
        __builtin_prefetch((const void*)(v1 + 32), 0, 1);

        o0 = wmma16(Av0, Bp, o0);
        o1 = wmma16(Av1, Bp, o1);
    }

    float inv = 1.0f / lsum;
    int mbase = lo ? 0 : 8;
#pragma unroll
    for (int j = 0; j < 8; ++j) {
        int c0 = hh * HDIM + mbase + j;
        int c1 = c0 + 16;
        aO[((size_t)b * DIMC + c0) * NN + qcol] = o0[j] * inv;
        aO[((size_t)b * DIMC + c1) * NN + qcol] = o1[j] * inv;
    }
}

// =====================================================================
// Kernel 4: output projection GEMM with fused (attn + lepe) and bias.
// out[b,o,n] = sum_c Wp[o,c] * (aO[b,c,n] + lepe[b,c,n]) + bias[o]
// =====================================================================
__global__ void __launch_bounds__(128)
proj_kernel(const float* __restrict__ aO, const float* __restrict__ lep,
            const float* __restrict__ wp, const float* __restrict__ bp,
            float* __restrict__ out) {
    int tile  = blockIdx.x * 4 + (threadIdx.x >> 5);  // [0, 2*16*144)
    int ntile = tile % (NN / 16); tile /= (NN / 16);
    int otile = tile % 16;
    int b     = tile / 16;
    int lane  = threadIdx.x & 31;
    int colq  = lane & 15;
    bool lo   = lane < 16;

    const float* wrow = wp + (size_t)(otile * 16 + colq) * DIMC;
    int ncol = ntile * 16 + colq;
    const float* ab = aO  + (size_t)b * DIMC * NN;
    const float* lb = lep + (size_t)b * DIMC * NN;

    v8f acc = {};
    for (int cb = 0; cb < DIMC; cb += 32) {
        int a0 = cb + (lo ? 0 : 16);
        int a1 = cb + (lo ? 8 : 24);
        v16h A;
#pragma unroll
        for (int j = 0; j < 8; ++j) {
            A[j]     = (_Float16)wrow[a0 + j];
            A[8 + j] = (_Float16)wrow[a1 + j];
        }
        int k0 = cb + (lo ? 0 : 16);
        v16h B;
#pragma unroll
        for (int j = 0; j < 16; ++j) {
            size_t off = (size_t)(k0 + j) * NN + ncol;
            B[j] = (_Float16)(ab[off] + lb[off]);
        }
        acc = wmma16(A, B, acc);
    }

    int mbase = lo ? 0 : 8;
#pragma unroll
    for (int j = 0; j < 8; ++j) {
        int o = otile * 16 + mbase + j;
        out[((size_t)b * DIMC + o) * NN + ncol] = acc[j] + bp[o];
    }
}

// =====================================================================
extern "C" void kernel_launch(void* const* d_in, const int* in_sizes, int n_in,
                              void* d_out, int out_size, void* d_ws, size_t ws_size,
                              hipStream_t stream) {
    const float* x      = (const float*)d_in[0];
    const float* w_qkv  = (const float*)d_in[1];
    const float* w_lepe = (const float*)d_in[2];
    const float* b_lepe = (const float*)d_in[3];
    const float* w_proj = (const float*)d_in[4];
    const float* b_proj = (const float*)d_in[5];
    float* out = (float*)d_out;

    // workspace carve (~21.2 MB total)
    const size_t nh  = (size_t)BATCH * HEADS * NN * HDIM;  // per-tensor f16 elems
    const size_t nf  = (size_t)BATCH * DIMC * NN;          // per-tensor f32 elems
    _Float16* qT = (_Float16*)d_ws;
    _Float16* kT = qT + nh;
    _Float16* vR = kT + nh;
    float*    vF = (float*)(vR + nh);
    float*    lp = vF + nf;
    float*    aO = lp + nf;

    // 1) qkv projection: 2*48*144 = 13824 wave-tiles, 4 waves/block
    qkv_kernel<<<BATCH * 48 * (NN / 16) / 4, 128, 0, stream>>>(
        x, w_qkv, qT, kT, vR, vF);

    // 2) lepe depthwise conv
    lepe_kernel<<<(BATCH * DIMC * NN + 255) / 256, 256, 0, stream>>>(
        vF, w_lepe, b_lepe, lp);

    // 3) attention: 16*144 = 2304 wave-tiles, 4 waves/block
    attn_kernel<<<BATCH * HEADS * (NN / 16) / 4, 128, 0, stream>>>(
        qT, kT, vR, aO);

    // 4) output projection: 2*16*144 = 4608 wave-tiles
    proj_kernel<<<BATCH * (DIMC / 16) * (NN / 16) / 4, 128, 0, stream>>>(
        aO, lp, w_proj, b_proj, out);
}
